// RecurrentCal_88845693485516
// MI455X (gfx1250) — matrix operand order repaired and checked by reading, not compiled, not run on hardware
//
#include <hip/hip_runtime.h>

// ---------------------------------------------------------------------------
// RecurrentCal for MI455X (gfx1250, wave32, WMMA f32_16x16x32_f16)
// Pipeline: layout/f16 prep -> 4x conv1x1 -> 4x conv5x5 (implicit GEMM)
//           -> 55 interleaved LSTM1/LSTM2 steps -> final 1x1 conv + permute.
// All matmuls via v_wmma_f32_16x16x32_f16, f32 accumulate.
// 32-bit per-lane offsets (saddr+voffset form); wave-uniform row guards;
// branch-free epilogues (clamped bias index + select instead of exec masking).
// ---------------------------------------------------------------------------

typedef __attribute__((ext_vector_type(16))) _Float16 v16h;
typedef __attribute__((ext_vector_type(8)))  _Float16 v8h;
typedef __attribute__((ext_vector_type(8)))  float    v8f;

#define LSEQ  55
#define BATCH 64
#define NPIX  169                  // 13*13
#define NSEQ  (BATCH*NPIX)         // 10816
#define PTOT  (LSEQ*NSEQ)          // 594880  (multiple of 16)
#define HPAD  128                  // hidden 100 padded to 128
#define NGP   512                  // 4 gates * 128

// ----------------------------- helpers -------------------------------------

__device__ __forceinline__ v8f zero8f() {
  v8f z;
#pragma unroll
  for (int i = 0; i < 8; ++i) z[i] = 0.0f;
  return z;
}

__device__ __forceinline__ v16h zero16h() {
  v16h z;
#pragma unroll
  for (int i = 0; i < 16; ++i) z[i] = (_Float16)0;
  return z;
}

// A operand (16x32 f16, MxK): lane L holds M = L&15; lane group g = L>>4.
// v16h elements e=0..15: K = 16g + e -> contiguous 32B at half-off 16g
__device__ __forceinline__ v16h load_a_frag(const _Float16* __restrict__ base,
                                            unsigned halfoff, int g) {
  const v8h lo = *(const v8h*)(base + halfoff + 16u * g);
  const v8h hi = *(const v8h*)(base + halfoff + 16u * g + 8u);
  v16h a;
#pragma unroll
  for (int e = 0; e < 8; ++e) { a[e] = lo[e]; a[8 + e] = hi[e]; }
  return a;
}

// B operand (32x16 f16, KxN), pre-swizzled in global:
// dst[(((kc*numNc)+nc)*32 + lane)*16 + e] = B[kc*32 + 16*(lane>>4) + e][nc*16 + (lane&15)]
__device__ __forceinline__ v16h load_b_frag(const _Float16* __restrict__ Bswz,
                                            int kc, int nc, int numNc, int lane) {
  unsigned off = ((unsigned)(kc * numNc + nc) * 32u + (unsigned)lane) * 16u;
  const v8h b0 = *(const v8h*)(Bswz + off);
  const v8h b1 = *(const v8h*)(Bswz + off + 8u);
  v16h b;
#pragma unroll
  for (int e = 0; e < 8; ++e) { b[e] = b0[e]; b[8 + e] = b1[e]; }
  return b;
}

__device__ __forceinline__ v8f wmma_f16(v16h a, v16h b, v8f c) {
  return __builtin_amdgcn_wmma_f32_16x16x32_f16(false, a, false, b,
                                                (short)0, c, false, false);
}

__device__ __forceinline__ float sigf(float x) {
  return 1.0f / (1.0f + __expf(-x));
}

// ----------------------------- prep kernels --------------------------------

// x [B,13,13,L,24] f32 -> A0 f16 [PTOT,32], row p = ((l*B+b)*13+i)*13+j, pad 0
__global__ void prep_x_kernel(const float* __restrict__ x,
                              _Float16* __restrict__ A0, int total) {
  for (int idx = blockIdx.x * blockDim.x + threadIdx.x; idx < total;
       idx += gridDim.x * blockDim.x) {
    int c = idx & 31;
    int p = idx >> 5;
    int l = p / NSEQ;
    int rem = p - l * NSEQ;
    int b = rem / NPIX;
    int rr = rem - b * NPIX;
    int i = rr / 13;
    int j = rr - i * 13;
    float v = 0.0f;
    if (c < 24)
      v = x[(unsigned)((((b * 13 + i) * 13 + j) * LSEQ + l) * 24 + c)];
    A0[(unsigned)idx] = (_Float16)v;
  }
}

// plain [Ksrc,Nsrc] f32 -> swizzled f16 [Kp,Np] (zero padded)
__global__ void swizzle_b_plain(const float* __restrict__ W,
                                _Float16* __restrict__ dst,
                                int Ksrc, int Nsrc, int Kp, int Np) {
  const int numNc = Np >> 4;
  const int total = Kp * Np;
  for (int idx = blockIdx.x * blockDim.x + threadIdx.x; idx < total;
       idx += gridDim.x * blockDim.x) {
    int e = idx & 15;
    int lane = (idx >> 4) & 31;
    int t2 = idx >> 9;                   // kc*numNc + nc
    int nc = t2 % numNc;
    int kc = t2 / numNc;
    int k = kc * 32 + ((lane >> 4) << 4) + e;
    int n = nc * 16 + (lane & 15);
    float v = (k < Ksrc && n < Nsrc) ? W[(unsigned)(k * Nsrc + n)] : 0.0f;
    dst[(unsigned)idx] = (_Float16)v;
  }
}

// 5x5 conv weight [25,30,30] f32 -> swizzled f16 [800,32] (per-tap K chunks)
__global__ void swizzle_b_conv5(const float* __restrict__ W,
                                _Float16* __restrict__ dst) {
  const int total = 800 * 32;            // Kp=800, Np=32, numNc=2
  for (int idx = blockIdx.x * blockDim.x + threadIdx.x; idx < total;
       idx += gridDim.x * blockDim.x) {
    int e = idx & 15;
    int lane = (idx >> 4) & 31;
    int t2 = idx >> 9;
    int nc = t2 & 1;
    int kc = t2 >> 1;
    int k = kc * 32 + ((lane >> 4) << 4) + e;
    int n = nc * 16 + (lane & 15);
    int tap = k >> 5;
    int cin = k & 31;
    float v = (cin < 30 && n < 30) ? W[(unsigned)((tap * 30 + cin) * 30 + n)]
                                   : 0.0f;
    dst[(unsigned)idx] = (_Float16)v;
  }
}

// LSTM weight slice [rows,400] (at row_off of src) -> swizzled f16 [Kp,512]
// columns gate-padded: n = gate*128 + wi, wi<100 valid
__global__ void swizzle_b_lstm(const float* __restrict__ W,
                               _Float16* __restrict__ dst,
                               int row_off, int rows, int Kp) {
  const int total = Kp * NGP;            // numNc = 32
  for (int idx = blockIdx.x * blockDim.x + threadIdx.x; idx < total;
       idx += gridDim.x * blockDim.x) {
    int e = idx & 15;
    int lane = (idx >> 4) & 31;
    int t2 = idx >> 9;
    int nc = t2 & 31;
    int kc = t2 >> 5;
    int k = kc * 32 + ((lane >> 4) << 4) + e;
    int n = nc * 16 + (lane & 15);
    int gate = n >> 7;
    int wi = n & 127;
    float v = (k < rows && wi < 100)
                  ? W[(unsigned)((row_off + k) * 400 + gate * 100 + wi)]
                  : 0.0f;
    dst[(unsigned)idx] = (_Float16)v;
  }
}

__global__ void fill_f16_zero(_Float16* __restrict__ p, int n) {
  for (int i = blockIdx.x * blockDim.x + threadIdx.x; i < n;
       i += gridDim.x * blockDim.x)
    p[(unsigned)i] = (_Float16)0;
}

__global__ void fill_f32_zero(float* __restrict__ p, int n) {
  for (int i = blockIdx.x * blockDim.x + threadIdx.x; i < n;
       i += gridDim.x * blockDim.x)
    p[(unsigned)i] = 0.0f;
}

// ----------------------------- conv kernels --------------------------------

// Branch-free bias fetch: clamp index into [0,29], select 0 for pad columns.
__device__ __forceinline__ float bias_sel(const float* __restrict__ bias,
                                          int n) {
  int nc = (n < 30) ? n : 29;        // keeps every lane's load in bounds
  float v = bias[nc];                // cndmask'd index, no exec divergence
  return (n < 30) ? v : 0.0f;
}

// C[M,32] = relu(A[M,32] x B[32,32] + bias); f16 in/out, f32 accumulate.
// 8 waves/WG, wave = 16 rows x 32 cols -> WG = 128 rows. M % 16 == 0 so the
// row guard is wave-uniform: no divergent epilogue.
__global__ __launch_bounds__(256) void conv1x1_wmma(
    const _Float16* __restrict__ A, const _Float16* __restrict__ Bswz,
    const float* __restrict__ bias, _Float16* __restrict__ Out, int M) {
  const int lane = threadIdx.x & 31;
  const int wave = threadIdx.x >> 5;
  const int g = lane >> 4;
  const int l16 = lane & 15;
  const int rowbase = (blockIdx.x * 8 + wave) * 16;
  if (rowbase >= M) return;            // uniform per wave
  const unsigned m = (unsigned)(rowbase + l16);

  v8f acc0 = zero8f(), acc1 = zero8f();
  v16h a = load_a_frag(A, m * 32u, g);
  v16h b0 = load_b_frag(Bswz, 0, 0, 2, lane);
  v16h b1 = load_b_frag(Bswz, 0, 1, 2, lane);
  acc0 = wmma_f16(a, b0, acc0);
  acc1 = wmma_f16(a, b1, acc1);

  const float bn0 = bias_sel(bias, l16);
  const float bn1 = bias_sel(bias, 16 + l16);
#pragma unroll
  for (int r = 0; r < 8; ++r) {
    unsigned mr = (unsigned)(rowbase + r + 8 * g);
    float v0 = acc0[r] + bn0; v0 = (v0 > 0.0f) ? v0 : 0.0f;
    float v1 = acc1[r] + bn1; v1 = (v1 > 0.0f) ? v1 : 0.0f;
    Out[mr * 32u + (unsigned)l16] = (_Float16)v0;
    Out[mr * 32u + 16u + (unsigned)l16] = (_Float16)v1;
  }
}

// 5x5 SAME conv as implicit GEMM: K = 25 taps * 32ch; per-tap spatial gather.
__global__ __launch_bounds__(256) void conv5x5_wmma(
    const _Float16* __restrict__ A, const _Float16* __restrict__ Bswz,
    const float* __restrict__ bias, _Float16* __restrict__ Out, int M) {
  const int lane = threadIdx.x & 31;
  const int wave = threadIdx.x >> 5;
  const int g = lane >> 4;
  const int l16 = lane & 15;
  const int rowbase = (blockIdx.x * 8 + wave) * 16;
  if (rowbase >= M) return;            // uniform per wave
  const int m = rowbase + l16;
  const int img = m / NPIX;
  const int rem = m - img * NPIX;
  const int i = rem / 13;
  const int j = rem - i * 13;

  v8f acc0 = zero8f(), acc1 = zero8f();
#pragma unroll
  for (int tap = 0; tap < 25; ++tap) {
    const int ty = tap / 5, tx = tap - (tap / 5) * 5;
    const int si = i + ty - 2, sj = j + tx - 2;
    const bool inb = (si >= 0) && (si < 13) && (sj >= 0) && (sj < 13);
    v16h a = zero16h();
    if (inb) {
      unsigned srow = (unsigned)(m + (ty - 2) * 13 + (tx - 2));
      a = load_a_frag(A, srow * 32u, g);
    }
    v16h b0 = load_b_frag(Bswz, tap, 0, 2, lane);
    v16h b1 = load_b_frag(Bswz, tap, 1, 2, lane);
    acc0 = wmma_f16(a, b0, acc0);
    acc1 = wmma_f16(a, b1, acc1);
  }

  const float bn0 = bias_sel(bias, l16);
  const float bn1 = bias_sel(bias, 16 + l16);
#pragma unroll
  for (int r = 0; r < 8; ++r) {
    unsigned mr = (unsigned)(rowbase + r + 8 * g);
    float v0 = acc0[r] + bn0; v0 = (v0 > 0.0f) ? v0 : 0.0f;
    float v1 = acc1[r] + bn1; v1 = (v1 > 0.0f) ? v1 : 0.0f;
    Out[mr * 32u + (unsigned)l16] = (_Float16)v0;
    Out[mr * 32u + 16u + (unsigned)l16] = (_Float16)v1;
  }
}

// ----------------------------- LSTM step -----------------------------------

// One timestep: z = Xt @ Bx + Hprev @ Bh + b (gate-padded 4x128 cols),
// then gates + cell update. 8 waves: wm in {0,1} (rows), wn in {0..3} (cols).
// Each wave: 16 rows x 32 hidden cols x 4 gates -> 8 accumulators.
// Grid is exact (NSEQ % 32 == 0): no row guards at all.
__global__ __launch_bounds__(256) void lstm_step_wmma(
    const _Float16* __restrict__ Xt, int xkc,         // #32-K chunks of Xt
    const _Float16* __restrict__ Hprev,
    float* __restrict__ C,
    const _Float16* __restrict__ Bx, const _Float16* __restrict__ Bh,
    const float* __restrict__ bias,
    _Float16* __restrict__ Hout) {
  const int lane = threadIdx.x & 31;
  const int wave = threadIdx.x >> 5;
  const int wm = wave & 1;
  const int wn = wave >> 1;                 // 0..3
  const int g = lane >> 4;
  const int l16 = lane & 15;
  const int rowbase = (blockIdx.x * 2 + wm) * 16;
  const unsigned m = (unsigned)(rowbase + l16);
  const unsigned XK = (unsigned)(xkc * 32);

  v8f acc[4][2];
#pragma unroll
  for (int gt = 0; gt < 4; ++gt)
#pragma unroll
    for (int s = 0; s < 2; ++s) acc[gt][s] = zero8f();

  // x contribution
  for (int kc = 0; kc < xkc; ++kc) {
    v16h a = load_a_frag(Xt, m * XK + (unsigned)(kc * 32), g);
#pragma unroll
    for (int gt = 0; gt < 4; ++gt)
#pragma unroll
      for (int s = 0; s < 2; ++s) {
        int nc = gt * 8 + wn * 2 + s;
        v16h b = load_b_frag(Bx, kc, nc, 32, lane);
        acc[gt][s] = wmma_f16(a, b, acc[gt][s]);
      }
  }
  // h contribution (Kp = 128 -> 4 chunks)
#pragma unroll
  for (int kc = 0; kc < 4; ++kc) {
    v16h a = load_a_frag(Hprev, m * (unsigned)HPAD + (unsigned)(kc * 32), g);
#pragma unroll
    for (int gt = 0; gt < 4; ++gt)
#pragma unroll
      for (int s = 0; s < 2; ++s) {
        int nc = gt * 8 + wn * 2 + s;
        v16h b = load_b_frag(Bh, kc, nc, 32, lane);
        acc[gt][s] = wmma_f16(a, b, acc[gt][s]);
      }
  }

  // gates + cell update (TF order i, j, f, o; forget bias 1.0)
#pragma unroll
  for (int s = 0; s < 2; ++s) {
    const int n = wn * 32 + s * 16 + l16;
    if (n < 100) {
      const float bi = bias[n];
      const float bj = bias[100 + n];
      const float bf = bias[200 + n] + 1.0f;
      const float bo = bias[300 + n];
#pragma unroll
      for (int r = 0; r < 8; ++r) {
        unsigned mr = (unsigned)(rowbase + r + 8 * g);
        unsigned off = mr * (unsigned)HPAD + (unsigned)n;
        float zi = acc[0][s][r] + bi;
        float zj = acc[1][s][r] + bj;
        float zf = acc[2][s][r] + bf;
        float zo = acc[3][s][r] + bo;
        float cp = C[off];
        float cn = sigf(zf) * cp + sigf(zi) * tanhf(zj);
        float h = sigf(zo) * tanhf(cn);
        C[off] = cn;
        Hout[off] = (_Float16)h;
      }
    } else {
      // keep pad columns exactly zero (they feed the next GEMM's A operand)
#pragma unroll
      for (int r = 0; r < 8; ++r) {
        unsigned mr = (unsigned)(rowbase + r + 8 * g);
        Hout[mr * (unsigned)HPAD + (unsigned)n] = (_Float16)0;
      }
    }
  }
}

// ----------------------------- final conv + permute ------------------------

// out[b,i,j,l] = dot(H2[l, b*169+i*13+j, 0:100], we) + be
__global__ void final_conv_kernel(const _Float16* __restrict__ H2,
                                  const float* __restrict__ we,
                                  const float* __restrict__ be,
                                  float* __restrict__ Out, int total) {
  int idx = blockIdx.x * blockDim.x + threadIdx.x;
  if (idx >= total) return;
  int l = idx % LSEQ;
  int t = idx / LSEQ;
  int j = t % 13; t /= 13;
  int i = t % 13;
  int b = t / 13;
  unsigned hoff =
      ((unsigned)l * (unsigned)NSEQ + (unsigned)(b * NPIX + i * 13 + j)) *
      (unsigned)HPAD;
  float s = 0.0f;
#pragma unroll
  for (int kk = 0; kk < 12; ++kk) {            // 96 of 100 taps, vectorized
    v8h h8 = *(const v8h*)(H2 + hoff + (unsigned)(kk * 8));
#pragma unroll
    for (int e = 0; e < 8; ++e) s += (float)h8[e] * we[kk * 8 + e];
  }
#pragma unroll
  for (int k = 96; k < 100; ++k) s += (float)H2[hoff + (unsigned)k] * we[k];
  Out[(unsigned)idx] = s + be[0];
}

// ----------------------------- host launcher -------------------------------

extern "C" void kernel_launch(void* const* d_in, const int* in_sizes, int n_in,
                              void* d_out, int out_size, void* d_ws,
                              size_t ws_size, hipStream_t stream) {
  (void)in_sizes; (void)n_in; (void)out_size; (void)ws_size;

  const float* x = (const float*)d_in[0];
  const float* w1[4] = {(const float*)d_in[1], (const float*)d_in[3],
                        (const float*)d_in[5], (const float*)d_in[7]};
  const float* b1[4] = {(const float*)d_in[2], (const float*)d_in[4],
                        (const float*)d_in[6], (const float*)d_in[8]};
  const float* w2[4] = {(const float*)d_in[9], (const float*)d_in[11],
                        (const float*)d_in[13], (const float*)d_in[15]};
  const float* b2[4] = {(const float*)d_in[10], (const float*)d_in[12],
                        (const float*)d_in[14], (const float*)d_in[16]};
  const float* lw1 = (const float*)d_in[17];
  const float* lb1 = (const float*)d_in[18];
  const float* lw2 = (const float*)d_in[19];
  const float* lb2 = (const float*)d_in[20];
  const float* we = (const float*)d_in[21];
  const float* be = (const float*)d_in[22];
  float* out = (float*)d_out;

  // --- workspace carve (256B aligned) ---
  char* wsp = (char*)d_ws;
  auto carve = [&](size_t bytes) -> char* {
    char* p = wsp;
    wsp += (bytes + 255) & ~(size_t)255;
    return p;
  };
  _Float16* actA = (_Float16*)carve((size_t)PTOT * 32 * 2);
  _Float16* actB = (_Float16*)carve((size_t)PTOT * 32 * 2);
  _Float16* Bw1[4];
  for (int i = 0; i < 4; ++i) Bw1[i] = (_Float16*)carve(32 * 32 * 2);
  _Float16* Bw5[4];
  for (int i = 0; i < 4; ++i) Bw5[i] = (_Float16*)carve(800 * 32 * 2);
  _Float16* Bx1 = (_Float16*)carve(32 * NGP * 2);
  _Float16* Bh1 = (_Float16*)carve(128 * NGP * 2);
  _Float16* Bx2 = (_Float16*)carve(128 * NGP * 2);
  _Float16* Bh2 = (_Float16*)carve(128 * NGP * 2);
  _Float16* hz = (_Float16*)carve((size_t)NSEQ * HPAD * 2);
  _Float16* h1a = (_Float16*)carve((size_t)NSEQ * HPAD * 2);
  _Float16* h1b = (_Float16*)carve((size_t)NSEQ * HPAD * 2);
  float* c1 = (float*)carve((size_t)NSEQ * HPAD * 4);
  float* c2 = (float*)carve((size_t)NSEQ * HPAD * 4);
  _Float16* H2 = (_Float16*)carve((size_t)LSEQ * NSEQ * HPAD * 2);

  // --- prep: layout + weight swizzles + state zero-init (every call) ---
  prep_x_kernel<<<8192, 256, 0, stream>>>(x, actA, PTOT * 32);
  const int cins[4] = {24, 30, 30, 30};
  for (int i = 0; i < 4; ++i)
    swizzle_b_plain<<<4, 256, 0, stream>>>(w1[i], Bw1[i], cins[i], 30, 32, 32);
  for (int i = 0; i < 4; ++i)
    swizzle_b_conv5<<<100, 256, 0, stream>>>(w2[i], Bw5[i]);
  swizzle_b_lstm<<<64, 256, 0, stream>>>(lw1, Bx1, 0, 30, 32);
  swizzle_b_lstm<<<256, 256, 0, stream>>>(lw1, Bh1, 30, 100, 128);
  swizzle_b_lstm<<<256, 256, 0, stream>>>(lw2, Bx2, 0, 100, 128);
  swizzle_b_lstm<<<256, 256, 0, stream>>>(lw2, Bh2, 100, 100, 128);
  fill_f16_zero<<<1024, 256, 0, stream>>>(hz, NSEQ * HPAD);
  fill_f32_zero<<<1024, 256, 0, stream>>>(c1, NSEQ * HPAD);
  fill_f32_zero<<<1024, 256, 0, stream>>>(c2, NSEQ * HPAD);

  // --- conv stack (ping-pong actA/actB) ---
  const int gridM = (PTOT + 127) / 128;  // 128 rows per WG
  _Float16* src = actA;
  _Float16* dst = actB;
  for (int i = 0; i < 4; ++i) {
    conv1x1_wmma<<<gridM, 256, 0, stream>>>(src, Bw1[i], b1[i], dst, PTOT);
    _Float16* t = src; src = dst; dst = t;
  }
  for (int i = 0; i < 4; ++i) {
    conv5x5_wmma<<<gridM, 256, 0, stream>>>(src, Bw5[i], b2[i], dst, PTOT);
    _Float16* t = src; src = dst; dst = t;
  }
  const _Float16* xs = src;  // [55][10816][32], time-major

  // --- interleaved LSTM1/LSTM2 steps ---
  const int gridL = NSEQ / 32;  // 338 WGs, 32 rows each
  const _Float16* h1prev = hz;
  const _Float16* h2prev = hz;
  for (int t = 0; t < LSEQ; ++t) {
    _Float16* h1out = (t & 1) ? h1b : h1a;
    lstm_step_wmma<<<gridL, 256, 0, stream>>>(
        xs + (size_t)t * NSEQ * 32, 1, h1prev, c1, Bx1, Bh1, lb1, h1out);
    _Float16* h2out = H2 + (size_t)t * NSEQ * HPAD;
    lstm_step_wmma<<<gridL, 256, 0, stream>>>(
        h1out, 4, h2prev, c2, Bx2, Bh2, lb2, h2out);
    h1prev = h1out;
    h2prev = h2out;
  }

  // --- final 1x1 conv + [B,13,13,L] permute ---
  final_conv_kernel<<<(PTOT + 255) / 256, 256, 0, stream>>>(H2, we, be, out,
                                                            PTOT);
}
